// LSTMVAE_56418690400372
// MI455X (gfx1250) — matrix operand order repaired and checked by reading, not compiled, not run on hardware
//
#include <hip/hip_runtime.h>
#include <hip/hip_bf16.h>
#include <stdint.h>

#define USE_TDM 1

typedef __attribute__((ext_vector_type(16))) __bf16       v16bf;
typedef __attribute__((ext_vector_type(8)))  float        v8f;
typedef __attribute__((ext_vector_type(4)))  unsigned int u32x4;
typedef __attribute__((ext_vector_type(8)))  int          i32x8;
typedef __attribute__((ext_vector_type(4)))  int          i32x4;
typedef __attribute__((ext_vector_type(4)))  unsigned int uint4v;

#define D_  512
#define L_  2
#define G_  64
#define B_  64
#define T_  512
#define FD  2048      // 4*D
#define KC  1024      // concat K = D(x) + D(h)
#define BD  (B_ * D_)

// ---------------------------------------------------------------------------
// Swizzled gate-matrix index (D-fragment order):
//   gatesSw[mt][jt][lane][r], lane = (j&15) + 16*((b>>3)&1), r = b&7, mt=b>>4
// ---------------------------------------------------------------------------
__device__ inline size_t gidx(int b, int j) {
    return ((((size_t)(b >> 4) * 128 + (j >> 4)) * 32) +
            ((j & 15) + (((b >> 3) & 1) << 4))) * 8 + (b & 7);
}

// ---------------------------------------------------------------------------
// Gate GEMM: gates[64][2048] = concat(A0,A1)[64][1024] @ Wcat[1024][2048]
// Weights pre-swizzled into B-fragment-major order:
//   Wsw[kc64][nt][s][lane][e]  (per 64-K chunk, per 16-N tile, 2 K-subchunks)
// so each block's chunk is a contiguous 16KB blob -> TDM tensor_load_to_lds.
// Block = 256 threads (8 waves): 64x128 output tile; wave (wm,wn) -> 16x64.
// ---------------------------------------------------------------------------
__global__ __launch_bounds__(256)
void k_gates(const float* __restrict__ A0, int ldA0,      // k in [0,512)
             const float* __restrict__ A1, int ldA1,      // k in [512,1024)
             const __bf16* __restrict__ Wsw,              // swizzled weights
             float* __restrict__ gatesSw)                 // swizzled gates
{
    __shared__ v16bf As[8 * 32];    // [s*4 + mt][lane]   8 KB (A fragments)
    __shared__ v16bf Bs[16 * 32];   // [ntl*2 + s][lane] 16 KB (B fragments)

    const int tid    = threadIdx.x;
    const int lane   = tid & 31;
    const int wave   = tid >> 5;
    const int wm     = wave & 3;            // M subtile
    const int wn     = wave >> 2;           // N strip (4 tiles of 16)
    const int ntBase = blockIdx.x * 8;      // first of 8 N-tiles for block

    v8f acc[4];
#pragma unroll
    for (int j = 0; j < 4; ++j)
#pragma unroll
        for (int r = 0; r < 8; ++r) acc[j][r] = 0.0f;

    __bf16* AsRaw = (__bf16*)As;

    for (int kc = 0; kc < KC / 64; ++kc) {
        // ---- stage A: 64 rows x 64 K, fp32 -> bf16, stored fragment-major.
        // Each thread converts 8 contiguous K values -> one 16B LDS store.
#pragma unroll
        for (int it = 0; it < 2; ++it) {
            const int g    = tid + it * 256;        // 0..511
            const int row  = g >> 3;
            const int col0 = (g & 7) * 8;
            const int kg   = kc * 64 + col0;
            const float* src = (kg < D_) ? (A0 + (size_t)row * ldA0 + kg)
                                         : (A1 + (size_t)row * ldA1 + (kg - D_));
            __bf16 tmp[8];
#pragma unroll
            for (int e = 0; e < 8; ++e) tmp[e] = (__bf16)src[e];
            const int s     = col0 >> 5;
            const int kk    = col0 & 31;
            const int half  = (kk >> 3) & 1;        // A frag: K = e+8*half+8*(e>=8)
            const int ebase = (kk >> 4) << 3;
            const int dlane = (row & 15) + (half << 4);
            const int mt    = row >> 4;
            uint4v u;
            __builtin_memcpy(&u, tmp, 16);
            *(uint4v*)&AsRaw[(((s << 2) + mt) * 32 + dlane) * 16 + ebase] = u;
        }

        // ---- stage B: contiguous 16 KB of pre-swizzled weights -> LDS
        const __bf16* wsrc = Wsw + (size_t)(kc * 128 + ntBase) * (2 * 32 * 16);
#if USE_TDM && __has_builtin(__builtin_amdgcn_tensor_load_to_lds)
        if (wave == 0) {  // one DMA per block, issued by wave 0 (EXEC ignored)
            const uint64_t ga  = (uint64_t)(uintptr_t)wsrc;
            const uint32_t lds = (uint32_t)(uintptr_t)&Bs[0];
            // D# group 0: count=1, lds_addr, global_addr[56:0], type=2
            u32x4 g0;
            g0[0] = 1u;
            g0[1] = lds;
            g0[2] = (uint32_t)ga;
            g0[3] = (uint32_t)((ga >> 32) & 0x1FFFFFFu) | (2u << 30);
            // D# group 1: mask=0, data_size=1 (2B), 1-D tensor/tile of 8192 elems
            i32x8 g1;
            g1[0] = (int)(1u << 16);                   // data_size = 2 bytes
            g1[1] = (int)((8192u & 0xFFFFu) << 16);    // tensor_dim0 lo16
            g1[2] = (int)((8192u >> 16) | (1u << 16)); // tensor_dim0 hi / dim1=1
            g1[3] = (int)((8192u & 0xFFFFu) << 16);    // tile_dim0 = 8192
            g1[4] = 1;                                 // tile_dim1=1, tile_dim2=0
            g1[5] = 8192;                              // tensor_dim0_stride
            g1[6] = 0;
            g1[7] = 0;
            i32x4 gz4 = {0, 0, 0, 0};
            i32x8 gz8 = {0, 0, 0, 0, 0, 0, 0, 0};
            (void)gz8;
#if defined(__clang_major__) && (__clang_major__ >= 23)
            __builtin_amdgcn_tensor_load_to_lds(g0, g1, gz4, gz4, gz8, 0);
#else
            __builtin_amdgcn_tensor_load_to_lds(g0, g1, gz4, gz4, 0);
#endif
            __builtin_amdgcn_s_wait_tensorcnt(0);
        }
#else
        {   // fallback: cooperative vectorized copy (1024 x 16B)
            const uint4v* s4 = (const uint4v*)wsrc;
            uint4v*       d4 = (uint4v*)&Bs[0];
#pragma unroll
            for (int e = 0; e < 4; ++e) d4[tid + e * 256] = s4[tid + e * 256];
        }
#endif
        if (kc + 1 < KC / 64)
            __builtin_prefetch(
                (const void*)(Wsw + (size_t)((kc + 1) * 128 + ntBase) * 1024), 0, 1);

        __syncthreads();

#pragma unroll
        for (int s = 0; s < 2; ++s) {
            v16bf a = As[((s << 2) + wm) * 32 + lane];   // one 32B LDS read
#pragma unroll
            for (int j = 0; j < 4; ++j) {
                v16bf b = Bs[(((wn << 2) + j) * 2 + s) * 32 + lane];
                acc[j] = __builtin_amdgcn_wmma_f32_16x16x32_bf16(
                    false, a, false, b, (short)0, acc[j], false, false);
            }
        }
        __syncthreads();
    }

    // store accumulators in D-fragment order: one contiguous 32B store each
#pragma unroll
    for (int j = 0; j < 4; ++j) {
        const int jt = ntBase + (wn << 2) + j;
        *(v8f*)&gatesSw[(((size_t)wm * 128 + jt) * 32 + lane) * 8] = acc[j];
    }
}

// ---------------------------------------------------------------------------
// LSTM cell pointwise update (PyTorch gate order i,f,g,o); reads swizzled gates
// ---------------------------------------------------------------------------
__global__ void k_cell(const float* __restrict__ gatesSw,
                       const float* __restrict__ bias,   // [2048] (bih+bhh)
                       float* __restrict__ c,            // [64][512] state
                       float* __restrict__ h,            // [64][512] state
                       float* __restrict__ extra,        // optional out slice
                       int extraLd)
{
    const int i = blockIdx.x * blockDim.x + threadIdx.x;  // 0..BD-1
    const int b = i >> 9;
    const int d = i & (D_ - 1);
    const float gi = gatesSw[gidx(b, d)]           + bias[d];
    const float gf = gatesSw[gidx(b, D_ + d)]      + bias[D_ + d];
    const float gg = gatesSw[gidx(b, 2 * D_ + d)]  + bias[2 * D_ + d];
    const float go = gatesSw[gidx(b, 3 * D_ + d)]  + bias[3 * D_ + d];
    const float si = 1.0f / (1.0f + __expf(-gi));
    const float sf = 1.0f / (1.0f + __expf(-gf));
    const float so = 1.0f / (1.0f + __expf(-go));
    const float cn = sf * c[i] + si * tanhf(gg);
    const float hn = so * tanhf(cn);
    c[i] = cn;
    h[i] = hn;
    if (extra) extra[(size_t)b * extraLd + d] = hn;
}

// ---------------------------------------------------------------------------
// Weight prep: fp32 Wih/Whh -> bf16, B-fragment-major swizzle:
//   flat i = ((((kc*128)+nt)*2 + s)*32 + lane)*16 + e
//   k = kc*64 + s*32 + 8*(lane>>4) + (e&7) + 16*(e>>3) ; j = nt*16 + (lane&15)
// ---------------------------------------------------------------------------
__global__ void k_prep_w(const float* __restrict__ Wih,   // [2048][512]
                         const float* __restrict__ Whh,   // [2048][512]
                         __bf16* __restrict__ Wsw)
{
    const int i    = blockIdx.x * blockDim.x + threadIdx.x;  // over KC*FD
    const int e    = i & 15;
    const int lane = (i >> 4) & 31;
    const int s    = (i >> 9) & 1;
    const int nt   = (i >> 10) & 127;
    const int kc   = i >> 17;
    const int k = kc * 64 + s * 32 + ((lane >> 4) << 3) + (e & 7) + ((e >> 3) << 4);
    const int j = nt * 16 + (lane & 15);
    const float v = (k < D_) ? Wih[(size_t)j * D_ + k]
                             : Whh[(size_t)j * D_ + (k - D_)];
    Wsw[i] = (__bf16)v;
}

__global__ void k_bias_sum(const float* __restrict__ bih,
                           const float* __restrict__ bhh,
                           float* __restrict__ bias) {
    const int i = blockIdx.x * blockDim.x + threadIdx.x;  // over FD
    bias[i] = bih[i] + bhh[i];
}

__global__ void k_bcast_prev(const float* __restrict__ initial,  // [512]
                             float* __restrict__ prev) {          // [64][512]
    const int i = blockIdx.x * blockDim.x + threadIdx.x;  // over BD
    prev[i] = initial[i & (D_ - 1)];
}

// ---------------------------------------------------------------------------
// VAE: mu/sig projection + reparameterization (small: fp32 dot products)
// ---------------------------------------------------------------------------
__global__ void k_vae_musig(const float* __restrict__ st,    // [L][B][D]
                            const float* __restrict__ muW,   // [L][G][D]
                            const float* __restrict__ mub,   // [L][G]
                            const float* __restrict__ sigW,  // [L][G][D]
                            const float* __restrict__ sigb,  // [L][G]
                            const float* __restrict__ eps,   // [L][B][G]
                            float* __restrict__ mu_out,
                            float* __restrict__ sig_out,
                            float* __restrict__ z)
{
    const int i  = blockIdx.x * blockDim.x + threadIdx.x;  // over L*B*G
    const int l  = i >> 12;            // B*G = 4096
    const int bg = i & 4095;
    const int b  = bg >> 6;
    const int g  = bg & (G_ - 1);
    const float* s  = st   + ((size_t)l * B_ + b) * D_;
    const float* wm = muW  + ((size_t)l * G_ + g) * D_;
    const float* ws = sigW + ((size_t)l * G_ + g) * D_;
    float am = 0.0f, as = 0.0f;
    for (int d = 0; d < D_; ++d) { const float sv = s[d]; am += sv * wm[d]; as += sv * ws[d]; }
    am += mub[l * G_ + g];
    as += sigb[l * G_ + g];
    mu_out[i]  = am;
    sig_out[i] = as;
    z[i] = am + eps[i] * as;
}

__global__ void k_vae_out(const float* __restrict__ z,     // [L][B][G]
                          const float* __restrict__ outW,  // [L][D][G]
                          const float* __restrict__ outb,  // [L][D]
                          float* __restrict__ dst)         // [L][B][D]
{
    const int i  = blockIdx.x * blockDim.x + threadIdx.x;  // over L*B*D
    const int l  = i >> 15;            // B*D = 32768
    const int bd = i & 32767;
    const int b  = bd >> 9;
    const int d  = bd & (D_ - 1);
    const float* zz = z + ((size_t)l * B_ + b) * G_;
    const float* w  = outW + ((size_t)l * D_ + d) * G_;
    float a = outb[l * D_ + d];
    for (int g = 0; g < G_; ++g) a += zz[g] * w[g];
    dst[i] = a;
}

// ---------------------------------------------------------------------------
// Host-side orchestration
// ---------------------------------------------------------------------------
extern "C" void kernel_launch(void* const* d_in, const int* in_sizes, int n_in,
                              void* d_out, int out_size, void* d_ws, size_t ws_size,
                              hipStream_t stream) {
    (void)in_sizes; (void)n_in; (void)out_size; (void)ws_size;

    const float* x       = (const float*)d_in[0];   // [B][T][D]
    const float* initial = (const float*)d_in[1];   // [1][D]
    const float* eWih    = (const float*)d_in[2];
    const float* eWhh    = (const float*)d_in[3];
    const float* ebih    = (const float*)d_in[4];
    const float* ebhh    = (const float*)d_in[5];
    const float* dWih    = (const float*)d_in[6];
    const float* dWhh    = (const float*)d_in[7];
    const float* dbih    = (const float*)d_in[8];
    const float* dbhh    = (const float*)d_in[9];
    const float* hmuW    = (const float*)d_in[10];
    const float* hmub    = (const float*)d_in[11];
    const float* hsigW   = (const float*)d_in[12];
    const float* hsigb   = (const float*)d_in[13];
    const float* houtW   = (const float*)d_in[14];
    const float* houtb   = (const float*)d_in[15];
    const float* cmuW    = (const float*)d_in[16];
    const float* cmub    = (const float*)d_in[17];
    const float* csigW   = (const float*)d_in[18];
    const float* csigb   = (const float*)d_in[19];
    const float* coutW   = (const float*)d_in[20];
    const float* coutb   = (const float*)d_in[21];
    const float* eps_h   = (const float*)d_in[22];
    const float* eps_c   = (const float*)d_in[23];

    float* out   = (float*)d_out;
    float* mu_h  = out + (size_t)B_ * T_ * D_;
    float* mu_c  = mu_h + (size_t)L_ * B_ * G_;
    float* sig_h = mu_c + (size_t)L_ * B_ * G_;
    float* sig_c = sig_h + (size_t)L_ * B_ * G_;

    // workspace carve-up (256B aligned)
    char* ws = (char*)d_ws;
    size_t off = 0;
    auto take = [&](size_t bytes) -> char* {
        char* p = ws + off;
        off += (bytes + 255) & ~(size_t)255;
        return p;
    };
    __bf16* encW   = (__bf16*)take(sizeof(__bf16) * L_ * KC * FD);   // 8 MB
    __bf16* decW   = (__bf16*)take(sizeof(__bf16) * L_ * KC * FD);   // 8 MB
    float* encBias = (float*)take(sizeof(float) * L_ * FD);
    float* decBias = (float*)take(sizeof(float) * L_ * FD);
    float* gates   = (float*)take(sizeof(float) * B_ * FD);
    float* enc_h   = (float*)take(sizeof(float) * L_ * BD);
    float* enc_c   = (float*)take(sizeof(float) * L_ * BD);
    float* dec_h   = (float*)take(sizeof(float) * L_ * BD);
    float* dec_c   = (float*)take(sizeof(float) * L_ * BD);
    float* zh      = (float*)take(sizeof(float) * L_ * B_ * G_);
    float* zc      = (float*)take(sizeof(float) * L_ * B_ * G_);
    float* prev    = (float*)take(sizeof(float) * BD);

    // ---- weight / bias prep (bf16 weights stay L2-resident afterwards)
    for (int l = 0; l < L_; ++l) {
        k_prep_w<<<(KC * FD) / 256, 256, 0, stream>>>(
            eWih + (size_t)l * FD * D_, eWhh + (size_t)l * FD * D_,
            encW + (size_t)l * KC * FD);
        k_prep_w<<<(KC * FD) / 256, 256, 0, stream>>>(
            dWih + (size_t)l * FD * D_, dWhh + (size_t)l * FD * D_,
            decW + (size_t)l * KC * FD);
        k_bias_sum<<<FD / 256, 256, 0, stream>>>(ebih + l * FD, ebhh + l * FD,
                                                 encBias + l * FD);
        k_bias_sum<<<FD / 256, 256, 0, stream>>>(dbih + l * FD, dbhh + l * FD,
                                                 decBias + l * FD);
    }
    hipMemsetAsync(enc_h, 0, sizeof(float) * L_ * BD, stream);
    hipMemsetAsync(enc_c, 0, sizeof(float) * L_ * BD, stream);

    const dim3 gGrid(FD / 128);  // 16 blocks per step GEMM

    // ---- encoder: time-major, both layers per step
    for (int t = 0; t < T_; ++t) {
        for (int l = 0; l < L_; ++l) {
            const float* A0  = (l == 0) ? (x + (size_t)t * D_) : (enc_h + 0 * BD);
            const int   ldA0 = (l == 0) ? (T_ * D_) : D_;
            k_gates<<<gGrid, 256, 0, stream>>>(
                A0, ldA0, enc_h + (size_t)l * BD, D_,
                encW + (size_t)l * KC * FD, gates);
            k_cell<<<BD / 256, 256, 0, stream>>>(
                gates, encBias + l * FD,
                enc_c + (size_t)l * BD, enc_h + (size_t)l * BD,
                nullptr, 0);
        }
    }

    // ---- VAE on final (h, c): mu/sig -> z -> decoder initial states
    k_vae_musig<<<(L_ * B_ * G_) / 256, 256, 0, stream>>>(
        enc_h, hmuW, hmub, hsigW, hsigb, eps_h, mu_h, sig_h, zh);
    k_vae_musig<<<(L_ * B_ * G_) / 256, 256, 0, stream>>>(
        enc_c, cmuW, cmub, csigW, csigb, eps_c, mu_c, sig_c, zc);
    k_vae_out<<<(L_ * BD) / 256, 256, 0, stream>>>(zh, houtW, houtb, dec_h);
    k_vae_out<<<(L_ * BD) / 256, 256, 0, stream>>>(zc, coutW, coutb, dec_c);
    k_bcast_prev<<<BD / 256, 256, 0, stream>>>(initial, prev);

    // ---- decoder: autoregressive, top-layer h fed back as next input
    for (int t = 0; t < T_; ++t) {
        for (int l = 0; l < L_; ++l) {
            const float* A0 = (l == 0) ? ((t == 0) ? prev : (dec_h + 1 * BD))
                                       : (dec_h + 0 * BD);
            k_gates<<<gGrid, 256, 0, stream>>>(
                A0, D_, dec_h + (size_t)l * BD, D_,
                decW + (size_t)l * KC * FD, gates);
            float* extra = (l == 1) ? (out + (size_t)t * D_) : nullptr;
            int    exLd  = (l == 1) ? (T_ * D_) : 0;
            k_cell<<<BD / 256, 256, 0, stream>>>(
                gates, decBias + l * FD,
                dec_c + (size_t)l * BD, dec_h + (size_t)l * BD,
                extra, exLd);
        }
    }
}
